// EnvironmentalContextAttention_9414568313185
// MI455X (gfx1250) — matrix operand order repaired and checked by reading, not compiled, not run on hardware
//
#include <hip/hip_runtime.h>
#include <hip/hip_bf16.h>
#include <hip/hip_fp16.h>

typedef _Float16 f16;
typedef __attribute__((ext_vector_type(16))) _Float16 v16h;
typedef __attribute__((ext_vector_type(8)))  _Float16 v8h;
typedef __attribute__((ext_vector_type(8)))  float    v8f;

#define BB 2
#define SS 2048
#define DD 1024
#define HH 16
#define EE 256
#define HDIM 64
#define MM (BB*SS)

static constexpr int OUT_F32    = 0;
static constexpr int OUT_HEAD   = 1;  // f16 [B,H,S,64]
static constexpr int OUT_HEAD_T = 2;  // f16 [B,H,64,S]

// ---------------- WMMA fragment helpers (ISA 7.12.2 layouts) ----------------
__device__ __forceinline__ v16h wmma_a_frag(const f16* p, int ld, int row0, int k0,
                                            int lrow, int half) {
  const f16* r = p + (size_t)(row0 + lrow) * ld + k0 + half * 8;
  union { v16h v; v8h h[2]; } u;
  u.h[0] = *(const v8h*)(r);
  u.h[1] = *(const v8h*)(r + 16);
  return u.v;
}
__device__ __forceinline__ v16h wmma_b_frag(const f16* p, int ld, int n0, int k0,
                                            int lrow, int half) {
  const f16* r = p + (size_t)(n0 + lrow) * ld + k0 + half * 8;
  union { v16h v; v8h h[2]; } u;
  u.h[0] = *(const v8h*)(r);
  u.h[1] = *(const v8h*)(r + 16);
  return u.v;
}
__device__ __forceinline__ v8f wmma16(v16h a, v16h b, v8f c) {
  return __builtin_amdgcn_wmma_f32_16x16x32_f16(false, a, false, b, (short)0, c,
                                                false, false);
}

// ---------------- conversion kernels ----------------
__global__ __launch_bounds__(256) void cvt_f32_to_f16(const float* __restrict__ src,
                                                      f16* __restrict__ dst, int n) {
  int i = blockIdx.x * blockDim.x + threadIdx.x;
  if (i < n) dst[i] = (f16)src[i];
}
__global__ __launch_bounds__(256) void cvt_transpose(const float* __restrict__ W,
                                                     f16* __restrict__ Wt,
                                                     int K, int N) {
  int i = blockIdx.x * blockDim.x + threadIdx.x;
  if (i < K * N) {
    int k = i / N, n = i % N;
    Wt[(size_t)n * K + k] = (f16)W[i];
  }
}

// ---------------- environmental gates (tiny) ----------------
__global__ __launch_bounds__(256) void gates_kernel(const float* __restrict__ env,
                                                    const float* __restrict__ We,
                                                    const float* __restrict__ be,
                                                    const float* __restrict__ Wm,
                                                    const float* __restrict__ bm,
                                                    float* __restrict__ gates) {
  __shared__ float t[BB][DD];
  int tid = threadIdx.x;
  for (int idx = tid; idx < BB * DD; idx += 256) {
    int b = idx >> 10, d = idx & (DD - 1);
    float acc = be[d];
    for (int e = 0; e < EE; ++e) acc += env[b * EE + e] * We[e * DD + d];
    t[b][d] = acc;
  }
  __syncthreads();
  if (tid < BB * HH) {
    int b = tid >> 4, h = tid & (HH - 1);
    float acc = bm[h];
    for (int d = 0; d < DD; ++d) acc += t[b][d] * Wm[d * HH + h];
    gates[b * HH + h] = 1.0f / (1.0f + __expf(-acc));
  }
}

// ---------------- generic WMMA GEMM: out = A(f16,[M,K]) @ Bt^T + bias ----------------
// wave computes a 32x64 output tile (2 M-subtiles x 4 N-subtiles), ping-pong
// double buffering. 8 wmma per 12 b128 loads per k-step. Requires Kdim % 64 == 0.
__global__ __launch_bounds__(256) void wmma_gemm_kernel(const f16* __restrict__ A,
                                                        const f16* __restrict__ Bt,
                                                        const float* __restrict__ bias,
                                                        void* __restrict__ out,
                                                        int Mdim, int Ndim, int Kdim,
                                                        int mode) {
  const int lane = threadIdx.x & 31;
  const int half = lane >> 4;
  const int lrow = lane & 15;
  const int wave = blockIdx.x * (blockDim.x >> 5) + (threadIdx.x >> 5);
  const int ntn  = Ndim >> 6;
  const int tm   = wave / ntn;   // 32-row tile index
  const int tn   = wave % ntn;   // 64-col tile index
  if (tm * 32 >= Mdim) return;

  v8f c[2][4] = {};
  // buffer 0: k = 0
  v16h a0[2], b0[4];
#pragma unroll
  for (int i = 0; i < 2; ++i) a0[i] = wmma_a_frag(A, Kdim, tm * 32 + i * 16, 0, lrow, half);
#pragma unroll
  for (int j = 0; j < 4; ++j) b0[j] = wmma_b_frag(Bt, Kdim, tn * 64 + j * 16, 0, lrow, half);

  for (int k0 = 0; k0 < Kdim; k0 += 64) {
    // prefetch ~3 steps ahead (global_prefetch_b8)
    if (k0 + 192 < Kdim) {
#pragma unroll
      for (int i = 0; i < 2; ++i)
        __builtin_prefetch(A + (size_t)(tm * 32 + i * 16 + lrow) * Kdim + k0 + 192, 0, 1);
#pragma unroll
      for (int j = 0; j < 4; ++j)
        __builtin_prefetch(Bt + (size_t)(tn * 64 + j * 16 + lrow) * Kdim + k0 + 192, 0, 1);
    }
    // fill buffer 1 with k0+32, then consume buffer 0
    v16h a1[2], b1[4];
#pragma unroll
    for (int i = 0; i < 2; ++i) a1[i] = wmma_a_frag(A, Kdim, tm * 32 + i * 16, k0 + 32, lrow, half);
#pragma unroll
    for (int j = 0; j < 4; ++j) b1[j] = wmma_b_frag(Bt, Kdim, tn * 64 + j * 16, k0 + 32, lrow, half);
#pragma unroll
    for (int i = 0; i < 2; ++i)
#pragma unroll
      for (int j = 0; j < 4; ++j) c[i][j] = wmma16(a0[i], b0[j], c[i][j]);

    // refill buffer 0 with k0+64 (if any), then consume buffer 1
    if (k0 + 64 < Kdim) {
#pragma unroll
      for (int i = 0; i < 2; ++i) a0[i] = wmma_a_frag(A, Kdim, tm * 32 + i * 16, k0 + 64, lrow, half);
#pragma unroll
      for (int j = 0; j < 4; ++j) b0[j] = wmma_b_frag(Bt, Kdim, tn * 64 + j * 16, k0 + 64, lrow, half);
    }
#pragma unroll
    for (int i = 0; i < 2; ++i)
#pragma unroll
      for (int j = 0; j < 4; ++j) c[i][j] = wmma16(a1[i], b1[j], c[i][j]);
  }

#pragma unroll
  for (int i = 0; i < 2; ++i) {
#pragma unroll
    for (int j = 0; j < 4; ++j) {
      const int ncol = tn * 64 + j * 16 + lrow;
      const float bv = bias ? bias[ncol] : 0.0f;
#pragma unroll
      for (int r = 0; r < 8; ++r) {
        const int mrow = tm * 32 + i * 16 + half * 8 + r;
        const float val = c[i][j][r] + bv;
        if (mode == OUT_F32) {
          ((float*)out)[(size_t)mrow * Ndim + ncol] = val;
        } else {
          const int b = mrow >> 11, s = mrow & (SS - 1);
          const int h = ncol >> 6, hd = ncol & (HDIM - 1);
          f16* o = (f16*)out;
          if (mode == OUT_HEAD)
            o[(((size_t)(b * HH + h)) * SS + s) * HDIM + hd] = (f16)val;
          else
            o[(((size_t)(b * HH + h)) * HDIM + hd) * SS + s] = (f16)val;
        }
      }
    }
  }
}

// ---------------- flash attention: per block = one (b,h), 64 query rows ----------------
__global__ __launch_bounds__(128) void flash_attn_kernel(const f16* __restrict__ Qh,
                                                         const f16* __restrict__ Kh,
                                                         const f16* __restrict__ Vt,
                                                         const float* __restrict__ gates,
                                                         f16* __restrict__ ctxh) {
  __shared__ __align__(16) f16 Psh[4][16][64];
  const int lane = threadIdx.x & 31;
  const int half = lane >> 4;
  const int lrow = lane & 15;
  const int w    = threadIdx.x >> 5;
  const int nqt  = SS / 64;
  const int bh   = blockIdx.x / nqt;
  const int qt   = blockIdx.x % nqt;
  const int b    = bh / HH, h = bh % HH;
  const f16* Qp = Qh + (size_t)bh * SS * HDIM;
  const f16* Kp = Kh + (size_t)bh * SS * HDIM;
  const f16* Vp = Vt + (size_t)bh * HDIM * SS;
  const int  qrow0 = qt * 64 + w * 16;
  const float scale = gates[bh] * 0.125f;  // gate / sqrt(64)

  const v16h aq0 = wmma_a_frag(Qp, HDIM, qrow0, 0,  lrow, half);
  const v16h aq1 = wmma_a_frag(Qp, HDIM, qrow0, 32, lrow, half);

  float mrun[8], lrun[8];
  v8f o[4] = {};
#pragma unroll
  for (int r = 0; r < 8; ++r) { mrun[r] = -3.0e38f; lrun[r] = 0.0f; }

  for (int kb = 0; kb < SS / 64; ++kb) {
    const int key0 = kb * 64;

    // ---- S = Q @ K^T (scaled) ----
    v16h bk[4][2];
#pragma unroll
    for (int j = 0; j < 4; ++j) {
      bk[j][0] = wmma_b_frag(Kp, HDIM, key0 + j * 16, 0,  lrow, half);
      bk[j][1] = wmma_b_frag(Kp, HDIM, key0 + j * 16, 32, lrow, half);
    }
    v8f sc[4] = {};
#pragma unroll
    for (int j = 0; j < 4; ++j) {
      sc[j] = wmma16(aq0, bk[j][0], sc[j]);
      sc[j] = wmma16(aq1, bk[j][1], sc[j]);
    }

    // ---- issue V fragment loads now; softmax VALU below hides their latency ----
    v16h bvf[2][4];
#pragma unroll
    for (int cc = 0; cc < 2; ++cc)
#pragma unroll
      for (int j = 0; j < 4; ++j)
        bvf[cc][j] = wmma_b_frag(Vp, SS, j * 16, key0 + cc * 32, lrow, half);

    // prefetch next key block's K and V rows
    if (key0 + 64 < SS) {
#pragma unroll
      for (int j = 0; j < 4; ++j) {
        __builtin_prefetch(Kp + (size_t)(key0 + 64 + j * 16 + lrow) * HDIM, 0, 1);
        __builtin_prefetch(Vp + (size_t)(j * 16 + lrow) * SS + key0 + 64, 0, 1);
      }
    }

    // ---- online softmax (row = half*8 + r, spread over 16 lanes) ----
#pragma unroll
    for (int r = 0; r < 8; ++r) {
      float m = -3.0e38f;
#pragma unroll
      for (int j = 0; j < 4; ++j) {
        sc[j][r] = sc[j][r] * scale;
        m = fmaxf(m, sc[j][r]);
      }
      m = fmaxf(m, __shfl_xor(m, 1, 32));
      m = fmaxf(m, __shfl_xor(m, 2, 32));
      m = fmaxf(m, __shfl_xor(m, 4, 32));
      m = fmaxf(m, __shfl_xor(m, 8, 32));
      const float mnew = fmaxf(mrun[r], m);
      const float corr = __expf(mrun[r] - mnew);
      mrun[r] = mnew;
      lrun[r] *= corr;
#pragma unroll
      for (int j = 0; j < 4; ++j) o[j][r] = o[j][r] * corr;
      float ps = 0.0f;
#pragma unroll
      for (int j = 0; j < 4; ++j) {
        const float p = __expf(sc[j][r] - mnew);
        ps += p;
        Psh[w][half * 8 + r][j * 16 + lrow] = (f16)p;
      }
      ps += __shfl_xor(ps, 1, 32);
      ps += __shfl_xor(ps, 2, 32);
      ps += __shfl_xor(ps, 4, 32);
      ps += __shfl_xor(ps, 8, 32);
      lrun[r] += ps;
    }

    // ---- O += P @ V (P via wave-private LDS transpose; V already in regs) ----
    const f16* Pp = &Psh[w][0][0];
#pragma unroll
    for (int cc = 0; cc < 2; ++cc) {
      v16h ap = wmma_a_frag(Pp, 64, 0, cc * 32, lrow, half);
#pragma unroll
      for (int j = 0; j < 4; ++j) o[j] = wmma16(ap, bvf[cc][j], o[j]);
    }
  }

  // normalize and write ctx as f16 [B,S,D] with head interleave
#pragma unroll
  for (int r = 0; r < 8; ++r) {
    const float inv = 1.0f / lrun[r];
    const int s = qrow0 + half * 8 + r;
#pragma unroll
    for (int j = 0; j < 4; ++j) {
      const int col = h * HDIM + j * 16 + lrow;
      ctxh[((size_t)(b * SS + s)) * DD + col] = (f16)(o[j][r] * inv);
    }
  }
}

// ---------------- launcher ----------------
extern "C" void kernel_launch(void* const* d_in, const int* in_sizes, int n_in,
                              void* d_out, int out_size, void* d_ws, size_t ws_size,
                              hipStream_t stream) {
  const float* x   = (const float*)d_in[0];
  const float* env = (const float*)d_in[1];
  const float* Wq  = (const float*)d_in[2];
  const float* bq  = (const float*)d_in[3];
  const float* Wk  = (const float*)d_in[4];
  const float* bk  = (const float*)d_in[5];
  const float* Wv  = (const float*)d_in[6];
  const float* bv  = (const float*)d_in[7];
  const float* Wo  = (const float*)d_in[8];
  const float* bo  = (const float*)d_in[9];
  const float* We  = (const float*)d_in[10];
  const float* be  = (const float*)d_in[11];
  const float* Wm  = (const float*)d_in[12];
  const float* bm  = (const float*)d_in[13];
  float* out = (float*)d_out;

  char* ws = (char*)d_ws;
  size_t off = 0;
  auto alloc = [&](size_t bytes) -> void* {
    off = (off + 255) & ~(size_t)255;
    void* p = ws + off;
    off += bytes;
    return p;
  };
  f16* xh   = (f16*)alloc((size_t)MM * DD * sizeof(f16));
  f16* WqT  = (f16*)alloc((size_t)DD * DD * sizeof(f16));
  f16* WkT  = (f16*)alloc((size_t)DD * DD * sizeof(f16));
  f16* WvT  = (f16*)alloc((size_t)DD * DD * sizeof(f16));
  f16* WoT  = (f16*)alloc((size_t)DD * DD * sizeof(f16));
  f16* Qh   = (f16*)alloc((size_t)BB * HH * SS * HDIM * sizeof(f16));
  f16* Kh   = (f16*)alloc((size_t)BB * HH * SS * HDIM * sizeof(f16));
  f16* Vt   = (f16*)alloc((size_t)BB * HH * HDIM * SS * sizeof(f16));
  f16* ctxh = (f16*)alloc((size_t)MM * DD * sizeof(f16));
  float* gates = (float*)alloc(BB * HH * sizeof(float));
  (void)ws_size; (void)in_sizes; (void)n_in; (void)out_size;

  // 1) precision conversion / weight transposition
  cvt_f32_to_f16<<<(MM * DD + 255) / 256, 256, 0, stream>>>(x, xh, MM * DD);
  cvt_transpose<<<(DD * DD + 255) / 256, 256, 0, stream>>>(Wq, WqT, DD, DD);
  cvt_transpose<<<(DD * DD + 255) / 256, 256, 0, stream>>>(Wk, WkT, DD, DD);
  cvt_transpose<<<(DD * DD + 255) / 256, 256, 0, stream>>>(Wv, WvT, DD, DD);
  cvt_transpose<<<(DD * DD + 255) / 256, 256, 0, stream>>>(Wo, WoT, DD, DD);

  // 2) environmental gates
  gates_kernel<<<1, 256, 0, stream>>>(env, We, be, Wm, bm, gates);

  // 3) Q/K/V projections (WMMA), head-major f16 outputs (V transposed)
  const int gemm_blocks = (MM / 32) * (DD / 64) / 8;  // 256
  wmma_gemm_kernel<<<gemm_blocks, 256, 0, stream>>>(xh, WqT, bq, Qh, MM, DD, DD, OUT_HEAD);
  wmma_gemm_kernel<<<gemm_blocks, 256, 0, stream>>>(xh, WkT, bk, Kh, MM, DD, DD, OUT_HEAD);
  wmma_gemm_kernel<<<gemm_blocks, 256, 0, stream>>>(xh, WvT, bv, Vt, MM, DD, DD, OUT_HEAD_T);

  // 4) gated flash attention
  flash_attn_kernel<<<BB * HH * (SS / 64), 128, 0, stream>>>(Qh, Kh, Vt, gates, ctxh);

  // 5) output projection (WMMA) -> f32 d_out
  wmma_gemm_kernel<<<gemm_blocks, 256, 0, stream>>>(ctxh, WoT, bo, out, MM, DD, DD, OUT_F32);
}
